// DynamicDeformableConv1d_26663156974178
// MI455X (gfx1250) — matrix operand order repaired and checked by reading, not compiled, not run on hardware
//
#include <hip/hip_runtime.h>

typedef __attribute__((ext_vector_type(16))) __bf16 v16bf;
typedef __attribute__((ext_vector_type(8)))  __bf16 v8bf;
typedef __attribute__((ext_vector_type(8)))  float  v8f;

#define CIN   256
#define COUT  256
#define LEN   4096
#define KTAP  3
#define LPAD  (LEN + 2)       // padded length (PAD=1 both sides)
#define LTILE 32              // output columns per block (2 WMMA N-tiles)
#define FRAGS 24              // u16 per B-fragment row (48B: 16B-aligned, bank-friendly)

// f32 -> bf16 via native conversion (RNE per ISA; lowers to v_cvt_*bf16_f32)
__device__ __forceinline__ unsigned short f2bfbits(float f) {
  return __builtin_bit_cast(unsigned short, (__bf16)f);
}

// ---------------------------------------------------------------------------
// Pack weights into WMMA A-fragment order, bf16:
//   wpack[ ((mtile*3 + kk)*8 + cc)*32 + lane ][ e=0..15 ]
// element pair (2vi,2vi+1) holds channels c0+kb, c0+kb+1 per ISA A layout:
//   kb = (vi<4?0:16) + (vi&3)*2 + half*8,  row m = mtile*16 + (lane&15)
// ---------------------------------------------------------------------------
__global__ void pack_weights(const float* __restrict__ weight,
                             unsigned* __restrict__ wpack_u32) {
  int t = blockIdx.x * 256 + threadIdx.x;
  if (t >= 16 * 3 * 8 * 32 * 8) return;
  const int vi   = t & 7;        t >>= 3;
  const int lane = t & 31;       t >>= 5;
  const int cc   = t & 7;        t >>= 3;
  const int kk   = t % 3;
  const int mtile = t / 3;

  const int half = lane >> 4;
  const int m    = mtile * 16 + (lane & 15);
  const int kb   = ((vi < 4) ? 0 : 16) + (vi & 3) * 2 + half * 8;
  const int c    = cc * 32 + kb;

  const float* wr = weight + ((size_t)m * CIN + c) * KTAP + kk;
  const unsigned b0 = f2bfbits(wr[0]);
  const unsigned b1 = f2bfbits(wr[KTAP]);       // next channel, same tap
  wpack_u32[((((size_t)mtile * 3 + kk) * 8 + cc) * 32 + lane) * 8 + vi] =
      b0 | (b1 << 16);
}

// ---------------------------------------------------------------------------
// Fused deformable conv1d: offset conv + bilinear gather + WMMA GEMM
// Block: 512 threads = 16 waves; each wave owns one Cout 16-row tile and
// two 16-column WMMA tiles (N-tile = 32), sharing one A fragment per chunk.
// ---------------------------------------------------------------------------
__global__ __launch_bounds__(512)
void deform_conv1d_wmma(const float* __restrict__ x,
                        const __bf16* __restrict__ wpack,
                        const float* __restrict__ bias,
                        const float* __restrict__ offw,
                        const float* __restrict__ offb,
                        float* __restrict__ out) {
  const int l0 = blockIdx.x * LTILE;
  const int b  = blockIdx.y;
  const int t  = threadIdx.x;

  __shared__ float          s_part[4 * 96];
  __shared__ int            s_i0[96], s_i1[96];        // clamped x indices
  __shared__ float          s_w0[96], s_w1[96];        // pad-zeroed lerp weights
  __shared__ unsigned short s_frag[2][2][32 * FRAGS];  // [buf][colgroup][row][elem]

  const float* xb = x + (size_t)b * CIN * LEN;

  // ---------------- Phase 1: offset conv (3 taps x 32 cols) ----------------
  if (t < 384) {
    const int o96 = t % 96;          // kk*32 + l
    const int sub = t / 96;          // 0..3 (4 partial reducers per output)
    const int kk  = o96 >> 5;
    const int l   = o96 & 31;
    const int lg  = l0 + l;
    float acc = 0.f;
    const int cbase = sub * 64;
    for (int ci = 0; ci < 64; ++ci) {
      const int c = cbase + ci;
      const float* xr = xb + (size_t)c * LEN;
      const float* wr = offw + ((size_t)kk * CIN + c) * 3;
      #pragma unroll
      for (int j = 0; j < 3; ++j) {
        const int li = lg + j - 1;                    // PAD = 1
        const float xv = (li >= 0 && li < LEN) ? xr[li] : 0.f;
        acc += xv * wr[j];
      }
    }
    s_part[sub * 96 + o96] = acc;
  }
  __syncthreads();
  if (t < 96) {
    const int kk = t >> 5;
    const int l  = t & 31;
    float off = offb[kk];
    #pragma unroll
    for (int sub = 0; sub < 4; ++sub) off += s_part[sub * 96 + t];
    float pos = (float)kk + off + (float)(l0 + l) + 1.0f;  // + l*stride + PAD
    pos = fminf(fmaxf(pos, 0.f), (float)(LPAD - 1));
    const int p0 = (int)floorf(pos);
    const int p1 = min(p0 + 1, LPAD - 1);
    const float w0 = (float)p1 - pos;
    const float w1 = pos - (float)p0;
    // pad handling folded into clamp + weight zeroing -> branch-free gather
    const bool v0 = (p0 >= 1) && (p0 <= LEN);
    const bool v1 = (p1 >= 1) && (p1 <= LEN);
    s_i0[t] = v0 ? (p0 - 1) : 0;
    s_i1[t] = v1 ? (p1 - 1) : 0;
    s_w0[t] = v0 ? w0 : 0.f;
    s_w1[t] = v1 ? w1 : 0.f;
  }
  __syncthreads();

  // ---------------- Phase 2: GEMM via bf16 WMMA ----------------
  const int lane  = t & 31;
  const int mtile = t >> 5;          // 0..15 -> Cout 16-row tile
  const int col   = lane & 15;
  const int half  = lane >> 4;

  // staging mapping: thread (sc = c_local 0..31, sl = l 0..15) writes one
  // sample per column group in B-fragment order:
  //   fragment row = ((c>>3)&1)*16 + l (== consumer lane), element = (c>>4)*8 + (c&7)
  const int sc   = t >> 4;
  const int sl   = t & 15;
  const int fidx = (((sc >> 3) & 1) * 16 + sl) * FRAGS + ((sc >> 4) * 8 + (sc & 7));

  const float* xrb = xb + (size_t)sc * LEN;   // channel base for this stager

  v8f acc0 = {};
  v8f acc1 = {};
  int buf = 0;

  for (int kk = 0; kk < KTAP; ++kk) {
    // gather params for this thread's two columns (sl and sl+16)
    const int   ia0 = s_i0[kk * 32 + sl];
    const int   ia1 = s_i1[kk * 32 + sl];
    const float wa0 = s_w0[kk * 32 + sl];
    const float wa1 = s_w1[kk * 32 + sl];
    const int   ib0 = s_i0[kk * 32 + sl + 16];
    const int   ib1 = s_i1[kk * 32 + sl + 16];
    const float wb0 = s_w0[kk * 32 + sl + 16];
    const float wb1 = s_w1[kk * 32 + sl + 16];

    const __bf16* wbase =
        wpack + (((size_t)mtile * 3 + kk) * 8) * 32 * 16 + (size_t)lane * 16;

    for (int cc = 0; cc < 8; ++cc) {
      // ---- stage two bilinear samples (branch-free) ----
      {
        const size_t coff = (size_t)(cc * 32) * LEN;
        const float xa0 = xrb[coff + ia0];
        const float xa1 = xrb[coff + ia1];
        const float ya0 = xrb[coff + ib0];
        const float ya1 = xrb[coff + ib1];
        s_frag[buf][0][fidx] = f2bfbits(xa0 * wa0 + xa1 * wa1);
        s_frag[buf][1][fidx] = f2bfbits(ya0 * wb0 + ya1 * wb1);
      }
      __syncthreads();   // single barrier covers both WMMAs (double-buffered LDS)

      // ---- A: pre-packed fragment, two coalesced b128 loads, reused twice ----
      const v16bf a = *(const v16bf*)(wbase + (size_t)cc * (32 * 16));

      // ---- B0/B1: this lane's fragment rows, 2x ds_load_b128 each ----
      const unsigned short* fr0 = &s_frag[buf][0][lane * FRAGS];
      const unsigned short* fr1 = &s_frag[buf][1][lane * FRAGS];
      const v8bf b0lo = *(const v8bf*)fr0;
      const v8bf b0hi = *(const v8bf*)(fr0 + 8);
      const v8bf b1lo = *(const v8bf*)fr1;
      const v8bf b1hi = *(const v8bf*)(fr1 + 8);
      v16bf bm0, bm1;
      #pragma unroll
      for (int i = 0; i < 8; ++i) {
        bm0[i] = b0lo[i]; bm0[8 + i] = b0hi[i];
        bm1[i] = b1lo[i]; bm1[8 + i] = b1hi[i];
      }

      acc0 = __builtin_amdgcn_wmma_f32_16x16x32_bf16(
          false, a, false, bm0, (short)0, acc0, false, false);
      acc1 = __builtin_amdgcn_wmma_f32_16x16x32_bf16(
          false, a, false, bm1, (short)0, acc1, false, false);

      buf ^= 1;
    }
  }

  // ---------------- store: D rows r + 8*half, cols col and col+16 ----------------
  float* orow = out + ((size_t)b * COUT + mtile * 16) * LEN + l0;
  #pragma unroll
  for (int r = 0; r < 8; ++r) {
    const int row = r + 8 * half;
    const float bv = bias[mtile * 16 + row];
    orow[(size_t)row * LEN + col]      = acc0[r] + bv;
    orow[(size_t)row * LEN + col + 16] = acc1[r] + bv;
  }
}

extern "C" void kernel_launch(void* const* d_in, const int* in_sizes, int n_in,
                              void* d_out, int out_size, void* d_ws, size_t ws_size,
                              hipStream_t stream) {
  const float* x    = (const float*)d_in[0];   // [B, Cin, L]
  const float* w    = (const float*)d_in[1];   // [Cout, Cin, 3]
  const float* bias = (const float*)d_in[2];   // [Cout]
  const float* offw = (const float*)d_in[3];   // [3, Cin, 3]
  const float* offb = (const float*)d_in[4];   // [3]
  float* out = (float*)d_out;                  // [B, Cout, L]

  // ws: 16*3*8*32*16 bf16 = 384 KB of packed A fragments
  unsigned* wpack_u32 = (unsigned*)d_ws;

  pack_weights<<<(16 * 3 * 8 * 32 * 8 + 255) / 256, 256, 0, stream>>>(w, wpack_u32);

  const int B = in_sizes[0] / (CIN * LEN);
  dim3 grid(LEN / LTILE, B);
  deform_conv1d_wmma<<<grid, 512, 0, stream>>>(
      x, (const __bf16*)d_ws, bias, offw, offb, out);
}
